// NerfModel_25795573580320
// MI455X (gfx1250) — matrix operand order repaired and checked by reading, not compiled, not run on hardware
//
#include <hip/hip_runtime.h>
#include <hip/hip_bf16.h>
#include <math.h>

typedef __attribute__((ext_vector_type(4)))  _Float16 v4h;
typedef __attribute__((ext_vector_type(8)))  _Float16 v8h;
typedef __attribute__((ext_vector_type(16))) _Float16 v16h;
typedef __attribute__((ext_vector_type(8)))  float    v8f;

#define NGRID  512
#define SCALE_ 1.5f

// ---- A-fragment loader: 16-bit A 16x32 layout (ISA 7.12.2) ----
// lane 0-15: M=lane, VGPR0..3 = K 0..7, VGPR4..7 = K 16..23
// lane 16-31: M=lane-16, K 8..15 and 24..31
__device__ inline v16h load_a_frag(const _Float16* src, int ldk, int kb, int lane) {
    int m    = lane & 15;
    int koff = (lane >> 4) * 8;
    const _Float16* p = src + m * ldk + kb + koff;
    union { v16h v; v8h h[2]; } u;
    u.h[0] = *(const v8h*)(p);        // K = kb+koff .. +7
    u.h[1] = *(const v8h*)(p + 16);   // K = kb+16+koff .. +7
    return u.v;
}

// ---- weight -> LDS converter into pre-swizzled B-fragment layout ----
// B 16-bit 32x16: lane n = l&15; lanes 0-15 hold K 0..7,16..23, lanes 16-31 hold K 8..15,24..31.
// Stored so a lane's 16 halves are contiguous: idx = (((kc*ntiles)+nt)*32 + l)*16 + h
__device__ inline void load_w_swz(_Float16* dst, const float* w,
                                  int Kreal, int Nreal, int kchunks, int ntiles,
                                  int tid, int nth) {
    int total = kchunks * ntiles * 512;
    for (int i = tid; i < total; i += nth) {
        int h    = i & 15;
        int l    = (i >> 4) & 31;
        int rest = i >> 9;
        int nt   = rest % ntiles;
        int kc   = rest / ntiles;
        int K = kc * 32 + (l >> 4) * 8 + (h >> 3) * 16 + (h & 7);
        int n = nt * 16 + (l & 15);
        float v = (K < Kreal && n < Nreal) ? w[K * Nreal + n] : 0.0f;
        dst[i] = (_Float16)v;
    }
}

// ---- bias + relu + store D (f32 C/D 16x16 layout: lane=col, 8 VGPRs=8 rows) ----
__device__ inline void store_relu(_Float16* dst, int ldn, int nt, int lane,
                                  v8f acc, const float* bias) {
    int n     = nt * 16 + (lane & 15);
    int mbase = (lane >> 4) * 8;
    float bv  = bias[n];
#pragma unroll
    for (int r = 0; r < 8; ++r) {
        float v = acc[r] + bv;
        v = v > 0.0f ? v : 0.0f;
        dst[(mbase + r) * ldn + n] = (_Float16)v;
    }
}

__global__ __launch_bounds__(128) void nerf_mlp_kernel(
    const float* __restrict__ x,  const float* __restrict__ dirs,
    const float* __restrict__ pxy, const float* __restrict__ pyz, const float* __restrict__ pxz,
    const float* __restrict__ w1, const float* __restrict__ b1,
    const float* __restrict__ w2, const float* __restrict__ b2,
    const float* __restrict__ w3, const float* __restrict__ b3,
    const float* __restrict__ w4, const float* __restrict__ b4,
    const float* __restrict__ w5, const float* __restrict__ b5,
    float* __restrict__ outC, float* __restrict__ outSigma, int Btot)
{
    // ---- shared weights (swizzled f16) ----
    __shared__ __align__(32) _Float16 sW1[3 * 4 * 512];   // 96x64
    __shared__ __align__(32) _Float16 sW2[2 * 1 * 512];   // 64x16
    __shared__ __align__(32) _Float16 sW3[2 * 4 * 512];   // 42(->64)x64
    __shared__ __align__(32) _Float16 sW4[2 * 4 * 512];   // 64x64
    __shared__ __align__(32) _Float16 sW5[2 * 1 * 512];   // 64x3(->16)
    __shared__ float sB1[64], sB2[16], sB3[64], sB4[64], sB5[16];
    // ---- per-wave activation staging ----
    __shared__ __align__(32) _Float16 sFeat[4][16 * 96];  // feat, later a1
    __shared__ __align__(32) _Float16 sH[4][16 * 64];     // h1, later a2
    __shared__ __align__(32) _Float16 sG[4][16 * 64];     // g (PE + h2 cols)
    __shared__ int sRow[4][64];                           // [0:16) xy-row [16:32) yz [32:48) xz [48:64) mask

    const int tid  = threadIdx.x;
    const int wv   = tid >> 5;
    const int lane = tid & 31;

    // ---- cooperative weight load + swizzle ----
    load_w_swz(sW1, w1, 96, 64, 3, 4, tid, 128);
    load_w_swz(sW2, w2, 64, 16, 2, 1, tid, 128);
    load_w_swz(sW3, w3, 42, 64, 2, 4, tid, 128);
    load_w_swz(sW4, w4, 64, 64, 2, 4, tid, 128);
    load_w_swz(sW5, w5, 64,  3, 2, 1, tid, 128);
    for (int i = tid; i < 64; i += 128) { sB1[i] = b1[i]; sB3[i] = b3[i]; sB4[i] = b4[i]; }
    for (int i = tid; i < 16; i += 128) { sB2[i] = b2[i]; sB5[i] = (i < 3) ? b5[i] : 0.0f; }
    __syncthreads();

    const int p0 = (blockIdx.x * 4 + wv) * 16;

    // ---- phase 1: indices, mask, positional encoding (lanes 0..15 = points) ----
    if (lane < 16) {
        int p    = p0 + lane;
        bool inb = p < Btot;
        int pp   = inb ? p : 0;
        float fx = x[pp * 3 + 0], fy = x[pp * 3 + 1], fz = x[pp * 3 + 2];
        int msk = (inb && fabsf(fx) < SCALE_ && fabsf(fy) < SCALE_ && fabsf(fz) < SCALE_) ? 1 : 0;
        int ix = (int)((fx * (1.0f / (2.0f * SCALE_)) + 0.5f) * (float)NGRID);
        int iy = (int)((fy * (1.0f / (2.0f * SCALE_)) + 0.5f) * (float)NGRID);
        int iz = (int)((fz * (1.0f / (2.0f * SCALE_)) + 0.5f) * (float)NGRID);
        ix = min(max(ix, 0), NGRID - 1);
        iy = min(max(iy, 0), NGRID - 1);
        iz = min(max(iz, 0), NGRID - 1);
        sRow[wv][lane]      = (ix * NGRID + iy) * 96;
        sRow[wv][16 + lane] = (iy * NGRID + iz) * 96;
        sRow[wv][32 + lane] = (ix * NGRID + iz) * 96;
        sRow[wv][48 + lane] = msk;

        // positional encoding of direction -> g row (cols 0..26), zero pad 42..63
        _Float16* gr = &sG[wv][lane * 64];
        float dx = dirs[pp * 3 + 0], dy = dirs[pp * 3 + 1], dz = dirs[pp * 3 + 2];
        gr[0] = (_Float16)dx; gr[1] = (_Float16)dy; gr[2] = (_Float16)dz;
#pragma unroll
        for (int l = 0; l < 4; ++l) {
            float fr = (float)(1 << l);
            float a0 = dx * fr, a1 = dy * fr, a2 = dz * fr;
            gr[3 + l * 6 + 0] = (_Float16)__sinf(a0);
            gr[3 + l * 6 + 1] = (_Float16)__sinf(a1);
            gr[3 + l * 6 + 2] = (_Float16)__sinf(a2);
            gr[3 + l * 6 + 3] = (_Float16)__cosf(a0);
            gr[3 + l * 6 + 4] = (_Float16)__cosf(a1);
            gr[3 + l * 6 + 5] = (_Float16)__cosf(a2);
        }
#pragma unroll
        for (int c = 42; c < 64; ++c) gr[c] = (_Float16)0.0f;
    }

    // ---- phase 2: coalesced plane gathers, feat = f_xy*f_yz*f_xz -> f16 LDS ----
    {
        int c4 = lane * 4;
        for (int m = 0; m < 16; ++m) {
            if (lane < 24) {
                int rxy = sRow[wv][m], ryz = sRow[wv][16 + m], rxz = sRow[wv][32 + m];
                float4 fa = *(const float4*)(pxy + rxy + c4);
                float4 fb = *(const float4*)(pyz + ryz + c4);
                float4 fc = *(const float4*)(pxz + rxz + c4);
                v4h r;
                r[0] = (_Float16)(fa.x * fb.x * fc.x);
                r[1] = (_Float16)(fa.y * fb.y * fc.y);
                r[2] = (_Float16)(fa.z * fb.z * fc.z);
                r[3] = (_Float16)(fa.w * fb.w * fc.w);
                *(v4h*)&sFeat[wv][m * 96 + c4] = r;
            }
        }
    }

    // ---- layer 1: feat[16x96] @ w1[96x64] + b1, relu -> sH ----
#pragma unroll
    for (int nt = 0; nt < 4; ++nt) {
        v8f acc = {};
#pragma unroll
        for (int kc = 0; kc < 3; ++kc) {
            v16h a = load_a_frag(sFeat[wv], 96, kc * 32, lane);
            v16h b = *(const v16h*)&sW1[((kc * 4) + nt) * 512 + lane * 16];
            acc = __builtin_amdgcn_wmma_f32_16x16x32_f16(false, a, false, b, (short)0, acc, false, false);
        }
        store_relu(sH[wv], 64, nt, lane, acc, sB1);
    }

    // ---- layer 2: h1[16x64] @ w2[64x16] + b2, relu -> sigma (col 15) + g cols 27..41 ----
    {
        v8f acc = {};
#pragma unroll
        for (int kc = 0; kc < 2; ++kc) {
            v16h a = load_a_frag(sH[wv], 64, kc * 32, lane);
            v16h b = *(const v16h*)&sW2[kc * 512 + lane * 16];
            acc = __builtin_amdgcn_wmma_f32_16x16x32_f16(false, a, false, b, (short)0, acc, false, false);
        }
        int n     = lane & 15;
        int mbase = (lane >> 4) * 8;
        float bv  = sB2[n];
#pragma unroll
        for (int r = 0; r < 8; ++r) {
            int m   = mbase + r;
            float v = acc[r] + bv;
            v = v > 0.0f ? v : 0.0f;
            if (n == 15) {
                int p = p0 + m;
                if (p < Btot) outSigma[p] = sRow[wv][48 + m] ? v : 0.0f;
            } else {
                sG[wv][m * 64 + 27 + n] = (_Float16)v;
            }
        }
    }

    // ---- layer 3: g[16x64] @ w3[64x64] + b3, relu -> sFeat (ld 64) ----
#pragma unroll
    for (int nt = 0; nt < 4; ++nt) {
        v8f acc = {};
#pragma unroll
        for (int kc = 0; kc < 2; ++kc) {
            v16h a = load_a_frag(sG[wv], 64, kc * 32, lane);
            v16h b = *(const v16h*)&sW3[((kc * 4) + nt) * 512 + lane * 16];
            acc = __builtin_amdgcn_wmma_f32_16x16x32_f16(false, a, false, b, (short)0, acc, false, false);
        }
        store_relu(sFeat[wv], 64, nt, lane, acc, sB3);
    }

    // ---- layer 4: a1[16x64] @ w4[64x64] + b4, relu -> sH ----
#pragma unroll
    for (int nt = 0; nt < 4; ++nt) {
        v8f acc = {};
#pragma unroll
        for (int kc = 0; kc < 2; ++kc) {
            v16h a = load_a_frag(sFeat[wv], 64, kc * 32, lane);
            v16h b = *(const v16h*)&sW4[((kc * 4) + nt) * 512 + lane * 16];
            acc = __builtin_amdgcn_wmma_f32_16x16x32_f16(false, a, false, b, (short)0, acc, false, false);
        }
        store_relu(sH[wv], 64, nt, lane, acc, sB4);
    }

    // ---- layer 5: a2[16x64] @ w5[64x3] + b5, sigmoid, mask -> c ----
    {
        v8f acc = {};
#pragma unroll
        for (int kc = 0; kc < 2; ++kc) {
            v16h a = load_a_frag(sH[wv], 64, kc * 32, lane);
            v16h b = *(const v16h*)&sW5[kc * 512 + lane * 16];
            acc = __builtin_amdgcn_wmma_f32_16x16x32_f16(false, a, false, b, (short)0, acc, false, false);
        }
        int n = lane & 15;
        if (n < 3) {
            int mbase = (lane >> 4) * 8;
            float bv  = sB5[n];
#pragma unroll
            for (int r = 0; r < 8; ++r) {
                int m = mbase + r;
                int p = p0 + m;
                if (p < Btot) {
                    float v = acc[r] + bv;
                    float s = 1.0f / (1.0f + __expf(-v));
                    outC[(size_t)p * 3 + n] = sRow[wv][48 + m] ? s : 0.0f;
                }
            }
        }
    }
}

extern "C" void kernel_launch(void* const* d_in, const int* in_sizes, int n_in,
                              void* d_out, int out_size, void* d_ws, size_t ws_size,
                              hipStream_t stream) {
    const float* x    = (const float*)d_in[0];
    const float* dirs = (const float*)d_in[1];
    const float* pxy  = (const float*)d_in[2];
    const float* pyz  = (const float*)d_in[3];
    const float* pxz  = (const float*)d_in[4];
    const float* w1 = (const float*)d_in[5];  const float* b1 = (const float*)d_in[6];
    const float* w2 = (const float*)d_in[7];  const float* b2 = (const float*)d_in[8];
    const float* w3 = (const float*)d_in[9];  const float* b3 = (const float*)d_in[10];
    const float* w4 = (const float*)d_in[11]; const float* b4 = (const float*)d_in[12];
    const float* w5 = (const float*)d_in[13]; const float* b5 = (const float*)d_in[14];

    int Btot = in_sizes[0] / 3;
    float* outC     = (float*)d_out;
    float* outSigma = outC + (size_t)Btot * 3;

    int blocks = (Btot + 63) / 64;  // 4 waves/block, 16 points/wave
    nerf_mlp_kernel<<<blocks, 128, 0, stream>>>(
        x, dirs, pxy, pyz, pxz,
        w1, b1, w2, b2, w3, b3, w4, b4, w5, b5,
        outC, outSigma, Btot);
}
